// MultiHeadedAttention_58548994179264
// MI455X (gfx1250) — matrix operand order repaired and checked by reading, not compiled, not run on hardware
//
#include <hip/hip_runtime.h>

// ---------------- problem constants ----------------
constexpr int Bb  = 2;
constexpr int Ss  = 2048;
constexpr int Hh  = 16;
constexpr int Dd  = 64;
constexpr int HID = 1024;          // Hh * Dd
constexpr int Mrows = Bb * Ss;     // 4096 rows for all projection GEMMs

typedef _Float16 v16h __attribute__((ext_vector_type(16)));
typedef _Float16 h8   __attribute__((ext_vector_type(8)));
typedef _Float16 h4   __attribute__((ext_vector_type(4)));
typedef float    v8f  __attribute__((ext_vector_type(8)));

// Load a 16-half WMMA fragment as two contiguous 16B chunks.
__device__ __forceinline__ v16h ld_frag2(const _Float16* p0, const _Float16* p1) {
  h8 lo = *(const h8*)p0;
  h8 hi = *(const h8*)p1;
  v16h r;
#pragma unroll
  for (int i = 0; i < 8; ++i) { r[i] = lo[i]; r[i + 8] = hi[i]; }
  return r;
}

__device__ __forceinline__ v8f wmma_f16(v16h a, v16h b, v8f c) {
  return __builtin_amdgcn_wmma_f32_16x16x32_f16(false, a, false, b, (short)0, c,
                                                false, false);
}

// ---------------- fp32 -> fp16 elementwise convert ----------------
__global__ __launch_bounds__(256) void cvt_f32_f16(const float* __restrict__ in,
                                                   _Float16* __restrict__ out) {
  int i = (blockIdx.x * blockDim.x + threadIdx.x) * 4;
  float4 v = *(const float4*)(in + i);
  h4 o = { (_Float16)v.x, (_Float16)v.y, (_Float16)v.z, (_Float16)v.w };
  *(h4*)(out + i) = o;
}

// ---------------- fp32 W[k][n] -> fp16 Wt[n][k] transpose ----------------
__global__ __launch_bounds__(256) void transpose_cvt(const float* __restrict__ W,
                                                     _Float16* __restrict__ Wt) {
  __shared__ float tile[32][33];
  int x  = blockIdx.x * 32 + threadIdx.x;   // n
  int y0 = blockIdx.y * 32 + threadIdx.y;   // k
#pragma unroll
  for (int i = 0; i < 32; i += 8)
    tile[threadIdx.y + i][threadIdx.x] = W[(size_t)(y0 + i) * HID + x];
  __syncthreads();
  int ox = blockIdx.y * 32 + threadIdx.x;   // k (contiguous in output row)
  int oy = blockIdx.x * 32 + threadIdx.y;   // n
#pragma unroll
  for (int i = 0; i < 32; i += 8)
    Wt[(size_t)(oy + i) * HID + ox] = (_Float16)tile[threadIdx.x][threadIdx.y + i];
}

// ---------- register-blocked WMMA GEMM: wave computes a 32x64 tile --------
// C[4096x1024] = A[4096x1024] * Wt^T + bias, K = 1024.
// Per k-step: 2 A frags + 4 B frags (12 x b128) feed 8 wmma (1.5 loads/wmma).
// mode 0: f16 out, Q/K layout  [((b*H+h)*S + s)*D + d]
// mode 1: f16 out, V transposed [((b*H+h)*D + d)*S + s]
// mode 2: f32 out, row-major   [m*HID + n]
__global__ __launch_bounds__(256) void gemm_wmma(const _Float16* __restrict__ A,
                                                 const _Float16* __restrict__ Wt,
                                                 const float* __restrict__ bias,
                                                 _Float16* __restrict__ out_h,
                                                 float* __restrict__ out_f,
                                                 int mode) {
  const int lane = threadIdx.x & 31;
  const int wid  = (blockIdx.x * blockDim.x + threadIdx.x) >> 5;
  const int tn = wid & 15;            // 1024/64 = 16 column super-tiles
  const int tm = wid >> 4;            // 4096/32 = 128 row super-tiles
  const int bm = tm * 32, bn = tn * 64;

  const int l15 = lane & 15;
  const int kb  = (lane < 16) ? 0 : 8;    // A fragment K base
  const int ko  = (lane < 16) ? 0 : 8;    // B fragment K base
  const int hs  = (lane < 16) ? 0 : 8;    // C fragment row half

  const _Float16* arow[2];
#pragma unroll
  for (int mi = 0; mi < 2; ++mi)
    arow[mi] = A + (size_t)(bm + mi * 16 + l15) * HID;
  const _Float16* wrow[4];
#pragma unroll
  for (int ni = 0; ni < 4; ++ni)
    wrow[ni] = Wt + (size_t)(bn + ni * 16 + l15) * HID;

  v8f acc[2][4];
#pragma unroll
  for (int mi = 0; mi < 2; ++mi)
#pragma unroll
    for (int ni = 0; ni < 4; ++ni) acc[mi][ni] = {};

  for (int k = 0; k < HID; k += 32) {
    v16h af[2], bf[4];
#pragma unroll
    for (int mi = 0; mi < 2; ++mi)
      af[mi] = ld_frag2(arow[mi] + k + kb, arow[mi] + k + kb + 16);
#pragma unroll
    for (int ni = 0; ni < 4; ++ni)
      bf[ni] = ld_frag2(wrow[ni] + k + ko, wrow[ni] + k + ko + 16);
#pragma unroll
    for (int mi = 0; mi < 2; ++mi)
#pragma unroll
      for (int ni = 0; ni < 4; ++ni)
        acc[mi][ni] = wmma_f16(af[mi], bf[ni], acc[mi][ni]);
  }

#pragma unroll
  for (int ni = 0; ni < 4; ++ni) {
    const int   n  = bn + ni * 16 + l15;
    const float bv = bias[n];
    const int   h  = n >> 6;
    const int   d  = n & 63;
#pragma unroll
    for (int mi = 0; mi < 2; ++mi) {
      const int rowb = bm + mi * 16;
      if (mode == 0) {
#pragma unroll
        for (int j = 0; j < 8; ++j) {
          int mm = rowb + j + hs;
          int b_ = mm >> 11, s = mm & (Ss - 1);
          out_h[(((size_t)b_ * Hh + h) * Ss + s) * Dd + d] =
              (_Float16)(acc[mi][ni][j] + bv);
        }
      } else if (mode == 1) {
        int b_ = rowb >> 11;
        int s0 = (rowb & (Ss - 1)) + hs;
        h8 pk;
#pragma unroll
        for (int j = 0; j < 8; ++j) pk[j] = (_Float16)(acc[mi][ni][j] + bv);
        *(h8*)(out_h + ((size_t)(b_ * Hh + h) * Dd + d) * Ss + s0) = pk;
      } else {
#pragma unroll
        for (int j = 0; j < 8; ++j) {
          int mm = rowb + j + hs;
          out_f[(size_t)mm * HID + n] = acc[mi][ni][j] + bv;
        }
      }
    }
  }
}

// ---------------- streaming (flash) attention ----------------
// one wave per 16-query tile; 32 keys per iteration; online softmax.
__global__ __launch_bounds__(256) void attn_wmma(const _Float16* __restrict__ q16,
                                                 const _Float16* __restrict__ k16,
                                                 const _Float16* __restrict__ vt,
                                                 _Float16* __restrict__ att) {
  __shared__ _Float16 lds_p[8][16 * 32];   // per-wave P tile (C-layout -> A-layout)

  const int lane  = threadIdx.x & 31;
  const int wslot = threadIdx.x >> 5;
  const int wid   = (blockIdx.x * blockDim.x + threadIdx.x) >> 5;
  const int bh    = wid >> 7;              // S/16 = 128 query tiles per head
  const int qt    = wid & 127;
  const int base_q = qt * 16;

  const int l15 = lane & 15;
  const int kb  = (lane < 16) ? 0 : 8;
  const int ko  = (lane < 16) ? 0 : 8;
  const int hs  = (lane < 16) ? 0 : 8;

  const _Float16* qrow = q16 + ((size_t)bh * Ss + base_q + l15) * Dd;
  const v16h aq0 = ld_frag2(qrow + kb,      qrow + kb + 16);
  const v16h aq1 = ld_frag2(qrow + 32 + kb, qrow + 32 + kb + 16);

  float rmax[8], rsum[8];
#pragma unroll
  for (int j = 0; j < 8; ++j) { rmax[j] = -INFINITY; rsum[j] = 0.f; }
  v8f o0 = {}, o1 = {}, o2 = {}, o3 = {};
  const float scale = 0.125f;              // 1/sqrt(64)
  _Float16* lp = &lds_p[wslot][0];

  for (int kk = 0; kk < Ss; kk += 32) {
    // --- scores: Q(16x64) . K(32x64)^T, two 16-key column tiles ---
    const _Float16* kr0 = k16 + ((size_t)bh * Ss + kk + l15) * Dd + ko;
    const _Float16* kr1 = kr0 + (size_t)16 * Dd;
    v16h b00 = ld_frag2(kr0,      kr0 + 16);
    v16h b01 = ld_frag2(kr0 + 32, kr0 + 48);
    v16h b10 = ld_frag2(kr1,      kr1 + 16);
    v16h b11 = ld_frag2(kr1 + 32, kr1 + 48);
    v8f s0 = {}, s1 = {};
    s0 = wmma_f16(aq0, b00, s0); s0 = wmma_f16(aq1, b01, s0);
    s1 = wmma_f16(aq0, b10, s1); s1 = wmma_f16(aq1, b11, s1);

    // --- online softmax, per owned row j ---
#pragma unroll
    for (int j = 0; j < 8; ++j) {
      float x0 = s0[j] * scale, x1 = s1[j] * scale;
      float mx = fmaxf(x0, x1);
#pragma unroll
      for (int off = 8; off >= 1; off >>= 1) mx = fmaxf(mx, __shfl_xor(mx, off, 32));
      float nm = fmaxf(rmax[j], mx);
      float al = __expf(rmax[j] - nm);
      rmax[j]  = nm;
      float p0 = __expf(x0 - nm), p1 = __expf(x1 - nm);
      float ps = p0 + p1;
#pragma unroll
      for (int off = 8; off >= 1; off >>= 1) ps += __shfl_xor(ps, off, 32);
      rsum[j] = rsum[j] * al + ps;
      o0[j] *= al; o1[j] *= al; o2[j] *= al; o3[j] *= al;
      int prow = j + hs;
      lp[prow * 32 + l15]      = (_Float16)p0;
      lp[prow * 32 + 16 + l15] = (_Float16)p1;
    }
    asm volatile("s_wait_dscnt 0" ::: "memory");   // P stores visible before reload

    // --- P (16x32) as A fragment, V (32x64) as 4 B fragments ---
    const _Float16* pr = lp + l15 * 32;
    v16h pa = ld_frag2(pr + kb, pr + kb + 16);

    const _Float16* vb = vt + ((size_t)bh * Dd + l15) * Ss + kk + ko;
    v16h bv0 = ld_frag2(vb,                   vb + 16);
    v16h bv1 = ld_frag2(vb + (size_t)16 * Ss, vb + (size_t)16 * Ss + 16);
    v16h bv2 = ld_frag2(vb + (size_t)32 * Ss, vb + (size_t)32 * Ss + 16);
    v16h bv3 = ld_frag2(vb + (size_t)48 * Ss, vb + (size_t)48 * Ss + 16);
    o0 = wmma_f16(pa, bv0, o0);
    o1 = wmma_f16(pa, bv1, o1);
    o2 = wmma_f16(pa, bv2, o2);
    o3 = wmma_f16(pa, bv3, o3);
  }

  // --- normalize and store merged (B, S, H*D) in f16 for the final GEMM ---
  const int b_ = bh >> 4, h = bh & 15;
#pragma unroll
  for (int j = 0; j < 8; ++j) {
    float inv = 1.f / rsum[j];
    int s = base_q + j + hs;
    size_t rbase = ((size_t)b_ * Ss + s) * HID + h * Dd;
    att[rbase +  0 + l15] = (_Float16)(o0[j] * inv);
    att[rbase + 16 + l15] = (_Float16)(o1[j] * inv);
    att[rbase + 32 + l15] = (_Float16)(o2[j] * inv);
    att[rbase + 48 + l15] = (_Float16)(o3[j] * inv);
  }
}

// ---------------- host-side orchestration ----------------
extern "C" void kernel_launch(void* const* d_in, const int* in_sizes, int n_in,
                              void* d_out, int out_size, void* d_ws, size_t ws_size,
                              hipStream_t stream) {
  const float* X  = (const float*)d_in[0];
  const float* Wq = (const float*)d_in[1];
  const float* Wk = (const float*)d_in[2];
  const float* Wv = (const float*)d_in[3];
  const float* Wo = (const float*)d_in[4];
  const float* Bq = (const float*)d_in[5];
  const float* Bk = (const float*)d_in[6];
  const float* Bv = (const float*)d_in[7];
  const float* Bo = (const float*)d_in[8];
  float* out = (float*)d_out;

  char* ws = (char*)d_ws;
  _Float16* x16 = (_Float16*)(ws);                 //  8 MiB  (4096x1024 f16)
  _Float16* wtq = (_Float16*)(ws + 8388608);       //  2 MiB each, transposed f16
  _Float16* wtk = (_Float16*)(ws + 10485760);
  _Float16* wtv = (_Float16*)(ws + 12582912);
  _Float16* wto = (_Float16*)(ws + 14680064);
  _Float16* q16 = (_Float16*)(ws + 16777216);      //  8 MiB [b][h][s][d]
  _Float16* k16 = (_Float16*)(ws + 25165824);      //  8 MiB [b][h][s][d]
  _Float16* vtw = (_Float16*)(ws + 33554432);      //  8 MiB [b][h][d][s]
  _Float16* att = (_Float16*)(ws + 41943040);      //  8 MiB [b][s][h*d]

  // 1. convert activations to f16
  cvt_f32_f16<<<(Mrows * HID) / 4 / 256, 256, 0, stream>>>(X, x16);

  // 2. convert + transpose weights to Wt[n][k] f16
  dim3 tb(32, 8), tg(HID / 32, HID / 32);
  transpose_cvt<<<tg, tb, 0, stream>>>(Wq, wtq);
  transpose_cvt<<<tg, tb, 0, stream>>>(Wk, wtk);
  transpose_cvt<<<tg, tb, 0, stream>>>(Wv, wtv);
  transpose_cvt<<<tg, tb, 0, stream>>>(Wo, wto);

  // 3. QKV projections (2048 waves = 256 blocks of 8 waves, 32x64 per wave)
  gemm_wmma<<<256, 256, 0, stream>>>(x16, wtq, Bq, q16, nullptr, 0);
  gemm_wmma<<<256, 256, 0, stream>>>(x16, wtk, Bk, k16, nullptr, 0);
  gemm_wmma<<<256, 256, 0, stream>>>(x16, wtv, Bv, vtw, nullptr, 1);

  // 4. streaming attention (4096 waves = 512 blocks)
  attn_wmma<<<512, 256, 0, stream>>>(q16, k16, vtw, att);

  // 5. output projection, fp32 result
  gemm_wmma<<<256, 256, 0, stream>>>(att, wto, Bo, nullptr, out, 2);
}
